// AttnWROPEX_53326313947194
// MI455X (gfx1250) — matrix operand order repaired and checked by reading, not compiled, not run on hardware
//
#include <hip/hip_runtime.h>

// ---------------------------------------------------------------------------
// MI455X (gfx1250) attention block.  B=2, T=2048, D=1024, H=16, C=64.
// ~69 GFLOP over ~50MB unique data -> compute bound -> bf16 WMMA
// (16x16x32, 8x the K-density of fp32 WMMA) with fp32 accumulation.
// Double-buffered LDS tiles, async global->LDS copies for K tiles
// (ASYNCcnt), global_prefetch for GEMM streams.  wave32 / WGP model.
// ---------------------------------------------------------------------------

typedef __attribute__((ext_vector_type(16))) __bf16 v16bf;
typedef __attribute__((ext_vector_type(8)))  __bf16 v8bf;
typedef __attribute__((ext_vector_type(4)))  __bf16 v4bf;
typedef __attribute__((ext_vector_type(8)))  float  v8f;

#define WMMA_BF16(a, b, c) \
  __builtin_amdgcn_wmma_f32_16x16x32_bf16(false, (a), false, (b), (short)0, (c), false, false)

#define WAIT_ASYNC0() asm volatile("s_wait_asynccnt 0x0" ::: "memory")
#define WAIT_DS0()    asm volatile("s_wait_dscnt 0x0" ::: "memory")

// A-fragment (16x32, row m = lane%16): K chunks [8h..8h+7] and [16+8h..16+8h+7]
__device__ inline v16bf load_frag_a(const __bf16* p, int hh) {
  union { v16bf v; v8bf q[2]; } u;
  u.q[0] = *(const v8bf*)(p + 8 * hh);
  u.q[1] = *(const v8bf*)(p + 8 * hh + 16);
  return u.v;
}
// B-fragment (32x16, col n = lane%16, stored transposed): K chunks [8h..8h+7] and [16+8h..16+8h+7]
__device__ inline v16bf load_frag_b(const __bf16* p, int hh) {
  union { v16bf v; v8bf q[2]; } u;
  u.q[0] = *(const v8bf*)(p + 8 * hh);
  u.q[1] = *(const v8bf*)(p + 8 * hh + 16);
  return u.v;
}

__device__ inline v4bf cvt4(const float* p) {
  const float4 v = *(const float4*)p;
  v4bf r; r[0] = (__bf16)v.x; r[1] = (__bf16)v.y; r[2] = (__bf16)v.z; r[3] = (__bf16)v.w;
  return r;
}
__device__ inline v4bf cvt4(const __bf16* p) { return *(const v4bf*)p; }

__device__ inline float rmax16(float v) {
  v = fmaxf(v, __shfl_xor(v, 1, 32));
  v = fmaxf(v, __shfl_xor(v, 2, 32));
  v = fmaxf(v, __shfl_xor(v, 4, 32));
  v = fmaxf(v, __shfl_xor(v, 8, 32));
  return v;
}
__device__ inline float rsum16(float v) {
  v += __shfl_xor(v, 1, 32);
  v += __shfl_xor(v, 2, 32);
  v += __shfl_xor(v, 4, 32);
  v += __shfl_xor(v, 8, 32);
  return v;
}

// ---------------------------------------------------------------------------
// GEMM  Out[M,N] = A[M,K] @ Bw[K,N] + bias  (fp32 out, bf16 WMMA core)
// Block: 256 threads = 8 waves. Tile 128(M) x 128(N), K step 32,
// double-buffered LDS: one barrier per K-step, 8 WMMA per wave per step.
// ---------------------------------------------------------------------------
template <typename AT>
__global__ __launch_bounds__(256) void gemm_bias_kernel(
    const AT* __restrict__ A, const float* __restrict__ Bw,
    const float* __restrict__ bias, float* __restrict__ Out,
    int M, int N, int K) {
  __shared__ alignas(16) __bf16 Asl[2][128][40];  // [m][k], +8 pad
  __shared__ alignas(16) __bf16 Bsl[2][128][40];  // transposed: [n][k]
  const int tid = threadIdx.x;
  const int lane = tid & 31, w = tid >> 5;
  const int lm = lane & 15, hh = lane >> 4;
  const int n0 = blockIdx.x * 128;
  const int m0 = blockIdx.y * 128;

  v8f acc[8] = {};

  auto load_tiles = [&](int bufi, int k0) {
    {  // A tile: 128 x 32
      int c4 = (tid & 7) * 4;
      int r = tid >> 3;  // 0..31
#pragma unroll
      for (int rr = 0; rr < 4; ++rr) {
        int row = r + 32 * rr;
        *(v4bf*)&Asl[bufi][row][c4] = cvt4(&A[(size_t)(m0 + row) * K + k0 + c4]);
      }
    }
    {  // B tile: 32 x 128, stored transposed
      int n4 = (tid & 31) * 4;
      int kk = tid >> 5;  // 0..7
#pragma unroll
      for (int pp = 0; pp < 4; ++pp) {
        int kr = kk + 8 * pp;
        v4bf v = cvt4(&Bw[(size_t)(k0 + kr) * N + n0 + n4]);
        Bsl[bufi][n4 + 0][kr] = v[0]; Bsl[bufi][n4 + 1][kr] = v[1];
        Bsl[bufi][n4 + 2][kr] = v[2]; Bsl[bufi][n4 + 3][kr] = v[3];
      }
    }
  };

  load_tiles(0, 0);
  __syncthreads();
  int buf = 0;
  for (int k0 = 0; k0 < K; k0 += 32) {
    if (k0 + 32 < K) load_tiles(buf ^ 1, k0 + 32);
    if (k0 + 64 < K) {  // L2 prefetch of the tile after next
      __builtin_prefetch(&A[(size_t)(m0 + (tid >> 1)) * K + k0 + 64 + (tid & 1) * 16], 0, 1);
      __builtin_prefetch(&Bw[(size_t)(k0 + 64 + (tid >> 3)) * N + n0 + (tid & 7) * 16], 0, 1);
    }
    v16bf a = load_frag_a(&Asl[buf][w * 16 + lm][0], hh);
#pragma unroll
    for (int nt = 0; nt < 8; ++nt) {
      v16bf bb = load_frag_b(&Bsl[buf][nt * 16 + lm][0], hh);
      acc[nt] = WMMA_BF16(a, bb, acc[nt]);
    }
    __syncthreads();
    buf ^= 1;
  }

#pragma unroll
  for (int nt = 0; nt < 8; ++nt) {
    int n = n0 + nt * 16 + lm;
    float bv = bias[n];
#pragma unroll
    for (int r = 0; r < 8; ++r) {
      int m = m0 + w * 16 + 8 * hh + r;
      Out[(size_t)m * N + n] = acc[nt][r] + bv;
    }
  }
}

// ---------------------------------------------------------------------------
// RoPE + repack: q,k get rotary (score scale 1/8 folded into q), q/k/v packed
// to bf16 [B,H,T,C].  One thread per (b,t,h,pair i<32).
// ---------------------------------------------------------------------------
__global__ __launch_bounds__(256) void rope_pack_kernel(
    const float* __restrict__ qf, const float* __restrict__ kvf,
    __bf16* __restrict__ Qh, __bf16* __restrict__ Kh, __bf16* __restrict__ Vh) {
  int idx = blockIdx.x * 256 + threadIdx.x;  // total = 2*2048*16*32
  int i = idx & 31;
  int h = (idx >> 5) & 15;
  int t = (idx >> 9) & 2047;
  int b = idx >> 20;
  // inv_freq = 10000^(-i/32);  ln(10000) = 9.2103403...
  float inv = __expf(-(float)i * (9.210340371976184f / 32.0f));
  float ang = (float)t * inv;
  float sn, cs;
  __sincosf(ang, &sn, &cs);
  size_t rowq = ((size_t)b * 2048 + t) * 1024 + h * 64 + i;
  size_t rowkv = ((size_t)b * 2048 + t) * 2048 + h * 64 + i;
  float q1 = qf[rowq], q2 = qf[rowq + 32];
  float k1 = kvf[rowkv], k2 = kvf[rowkv + 32];
  float v1 = kvf[rowkv + 1024], v2 = kvf[rowkv + 1024 + 32];
  size_t po = ((size_t)(b * 16 + h) * 2048 + t) * 64 + i;
  const float qs = 0.125f;  // 1/sqrt(C)
  Qh[po]      = (__bf16)((q1 * cs - q2 * sn) * qs);
  Qh[po + 32] = (__bf16)((q2 * cs + q1 * sn) * qs);
  Kh[po]      = (__bf16)(k1 * cs - k2 * sn);
  Kh[po + 32] = (__bf16)(k2 * cs + k1 * sn);
  Vh[po]      = (__bf16)v1;
  Vh[po + 32] = (__bf16)v2;
}

// ---------------------------------------------------------------------------
// Flash-style attention per (b,h): block = 128 query rows (8 waves x 16 rows),
// online softmax over 64-key tiles, double-buffered.  K tiles are copied
// global->LDS with async-load instructions (ASYNCcnt); V tiles go through
// VGPRs (need transposition for the B-fragment layout).  16 WMMA / tile.
// ---------------------------------------------------------------------------
__global__ __launch_bounds__(256) void attn_kernel(
    const __bf16* __restrict__ Qh, const __bf16* __restrict__ Kh,
    const __bf16* __restrict__ Vh, __bf16* __restrict__ Ao) {
  __shared__ alignas(16) __bf16 Ksl[2][64][72];    // [key][c]   (B-layout for Q@K^T)
  __shared__ alignas(16) __bf16 Vtl[2][64][72];    // [c][key]   (B-layout for P@V)
  __shared__ alignas(16) __bf16 Psl[8][16][72];    // per-wave P staging [m][key]
  const int tid = threadIdx.x, lane = tid & 31, w = tid >> 5;
  const int lm = lane & 15, hh = lane >> 4;
  const int h = blockIdx.y, b = blockIdx.z;
  const int q0 = blockIdx.x * 128;
  const size_t bh = (size_t)(b * 16 + h);

  const __bf16* Qrow = Qh + (bh * 2048 + q0 + w * 16 + lm) * 64;
  v16bf qa0 = load_frag_a(Qrow, hh);       // c 0..31
  v16bf qa1 = load_frag_a(Qrow + 32, hh);  // c 32..63

  v8f o[4] = {};
  float mr[8], lr[8];
#pragma unroll
  for (int r = 0; r < 8; ++r) { mr[r] = -1e30f; lr[r] = 0.0f; }

  auto load_kv = [&](int bufi, int kt) {
    {  // K tile 64x64 bf16: pure byte copy -> async global->LDS (2x b128/thread)
      int key = tid >> 2;           // 0..63
      int c8 = (tid & 3) * 16;      // 0,16,32,48
      const __bf16* g = Kh + (bh * 2048 + kt + key) * 64 + c8;
      unsigned int l = (unsigned int)(uintptr_t)&Ksl[bufi][key][c8];
      asm volatile("global_load_async_to_lds_b128 %0, %1, off"
                   :: "v"(l), "v"((unsigned long long)(uintptr_t)g) : "memory");
      asm volatile("global_load_async_to_lds_b128 %0, %1, off"
                   :: "v"(l + 16u), "v"((unsigned long long)((uintptr_t)g + 16)) : "memory");
    }
    {  // V tile 64x64, transposed into [c][key] through VGPRs
      int c4 = (tid & 15) * 4;
      int key = tid >> 4;  // 0..15
#pragma unroll
      for (int pp = 0; pp < 4; ++pp) {
        int kk = key + 16 * pp;
        v4bf vv = *(const v4bf*)(Vh + (bh * 2048 + kt + kk) * 64 + c4);
        Vtl[bufi][c4 + 0][kk] = vv[0]; Vtl[bufi][c4 + 1][kk] = vv[1];
        Vtl[bufi][c4 + 2][kk] = vv[2]; Vtl[bufi][c4 + 3][kk] = vv[3];
      }
    }
  };

  load_kv(0, 0);
  WAIT_ASYNC0();
  __syncthreads();

  int buf = 0;
  for (int kt = 0; kt < 2048; kt += 64) {
    if (kt + 64 < 2048) load_kv(buf ^ 1, kt + 64);  // overlap with compute below

    // S = Q @ K^T over 64 keys (scale folded into Q): 8 WMMA
    v8f z = {};
    v8f s[4];
#pragma unroll
    for (int nt = 0; nt < 4; ++nt) {
      s[nt] = WMMA_BF16(qa0, load_frag_b(&Ksl[buf][nt * 16 + lm][0], hh), z);
      s[nt] = WMMA_BF16(qa1, load_frag_b(&Ksl[buf][nt * 16 + lm][32], hh), s[nt]);
    }

    // online softmax per query row (rows live in 16-lane half groups)
#pragma unroll
    for (int r = 0; r < 8; ++r) {
      float rm = rmax16(fmaxf(fmaxf(s[0][r], s[1][r]), fmaxf(s[2][r], s[3][r])));
      float nm = fmaxf(mr[r], rm);
      float alpha = __expf(mr[r] - nm);
      float e0 = __expf(s[0][r] - nm);
      float e1 = __expf(s[1][r] - nm);
      float e2 = __expf(s[2][r] - nm);
      float e3 = __expf(s[3][r] - nm);
      lr[r] = lr[r] * alpha + rsum16((e0 + e1) + (e2 + e3));
      mr[r] = nm;
#pragma unroll
      for (int nt = 0; nt < 4; ++nt) o[nt][r] *= alpha;
      int mrow = 8 * hh + r;
      Psl[w][mrow][lm]      = (__bf16)e0;
      Psl[w][mrow][16 + lm] = (__bf16)e1;
      Psl[w][mrow][32 + lm] = (__bf16)e2;
      Psl[w][mrow][48 + lm] = (__bf16)e3;
    }

    // per-wave LDS round-trip: D-layout P -> A-layout fragments (keys 0..63)
    WAIT_DS0();
    v16bf pa0 = load_frag_a(&Psl[w][lm][0], hh);
    v16bf pa1 = load_frag_a(&Psl[w][lm][32], hh);
#pragma unroll
    for (int nt = 0; nt < 4; ++nt) {  // O += P @ V : 8 WMMA
      o[nt] = WMMA_BF16(pa0, load_frag_b(&Vtl[buf][nt * 16 + lm][0], hh), o[nt]);
      o[nt] = WMMA_BF16(pa1, load_frag_b(&Vtl[buf][nt * 16 + lm][32], hh), o[nt]);
    }

    WAIT_ASYNC0();   // next K tile fully landed in LDS
    __syncthreads(); // everyone done reading `buf`, writes to buf^1 visible
    buf ^= 1;
  }

  // normalize and write bf16 [B,T,H*C] for the output projection GEMM
#pragma unroll
  for (int r = 0; r < 8; ++r) {
    float inv = 1.0f / lr[r];
    int t = q0 + w * 16 + 8 * hh + r;
    size_t base = ((size_t)b * 2048 + t) * 1024 + h * 64;
#pragma unroll
    for (int nt = 0; nt < 4; ++nt)
      Ao[base + nt * 16 + lm] = (__bf16)(o[nt][r] * inv);
  }
}

// ---------------------------------------------------------------------------
extern "C" void kernel_launch(void* const* d_in, const int* in_sizes, int n_in,
                              void* d_out, int out_size, void* d_ws, size_t ws_size,
                              hipStream_t stream) {
  const float* x   = (const float*)d_in[0];
  const float* Wq  = (const float*)d_in[1];
  const float* bq  = (const float*)d_in[2];
  const float* Wkv = (const float*)d_in[3];
  const float* bkv = (const float*)d_in[4];
  const float* Wo  = (const float*)d_in[5];
  const float* bo  = (const float*)d_in[6];
  float* out = (float*)d_out;

  char* ws = (char*)d_ws;
  float*  qf  = (float*)(ws);                          // 16 MB  q fp32
  float*  kvf = (float*)(ws + (size_t)(16u << 20));    // 32 MB  kv fp32
  __bf16* Qh  = (__bf16*)(ws + (size_t)(48u << 20));   //  8 MB  [B,H,T,C]
  __bf16* Kh  = (__bf16*)(ws + (size_t)(56u << 20));   //  8 MB
  __bf16* Vh  = (__bf16*)(ws + (size_t)(64u << 20));   //  8 MB
  __bf16* Ao  = (__bf16*)(ws + (size_t)(72u << 20));   //  8 MB  [B,T,H*C]

  dim3 blk(256);
  // q = x @ Wq + bq          (4096 x 1024 x 1024)
  gemm_bias_kernel<float><<<dim3(1024 / 128, 4096 / 128), blk, 0, stream>>>(
      x, Wq, bq, qf, 4096, 1024, 1024);
  // kv = x @ Wkv + bkv       (4096 x 2048 x 1024)
  gemm_bias_kernel<float><<<dim3(2048 / 128, 4096 / 128), blk, 0, stream>>>(
      x, Wkv, bkv, kvf, 4096, 2048, 1024);
  // RoPE + bf16 head packing
  rope_pack_kernel<<<dim3((2 * 2048 * 16 * 32) / 256), blk, 0, stream>>>(
      qf, kvf, Qh, Kh, Vh);
  // attention per (b,h)
  attn_kernel<<<dim3(2048 / 128, 16, 2), blk, 0, stream>>>(Qh, Kh, Vh, Ao);
  // out = attn_out @ Wo + bo (4096 x 1024 x 1024)
  gemm_bias_kernel<__bf16><<<dim3(1024 / 128, 4096 / 128), blk, 0, stream>>>(
      Ao, Wo, bo, out, 4096, 1024, 1024);
}
